// VQVAE_8658654068982
// MI455X (gfx1250) — matrix operand order repaired and checked by reading, not compiled, not run on hardware
//
#include <hip/hip_runtime.h>

typedef __attribute__((ext_vector_type(16))) __bf16        v16bf;
typedef __attribute__((ext_vector_type(8)))  float          v8f;
typedef __attribute__((ext_vector_type(16))) unsigned short u16x16;
typedef __attribute__((ext_vector_type(8)))  unsigned short u16x8;
typedef __attribute__((ext_vector_type(4)))  unsigned short u16x4;
typedef __attribute__((ext_vector_type(4)))  unsigned int   u32x4;
typedef __attribute__((ext_vector_type(8)))  int            i32x8;
typedef __attribute__((ext_vector_type(4)))  int            i32x4;

#define ROWS_PER_WAVE   16
#define ROWS_PER_BLOCK  128   // 8 waves * 16 rows
#define KCHUNK          128   // codes staged in LDS per chunk (TDM tile)
#define DDIM            64
#define LDS_STRIDE      72    // ushorts per code row (64 + 8 pad) = 144B, 16B aligned

// sched_group_barrier masks
#define SGB_VALU   0x002
#define SGB_MFMA   0x008   // matrix pipe (WMMA on gfx1250)
#define SGB_DSREAD 0x100

__device__ __forceinline__ unsigned short f2bf_rne(float x) {
  unsigned u = __builtin_bit_cast(unsigned, x);
  u += 0x7FFFu + ((u >> 16) & 1u);
  return (unsigned short)(u >> 16);
}
__device__ __forceinline__ float bf2f(unsigned short h) {
  return __builtin_bit_cast(float, (unsigned)h << 16);
}

__device__ __forceinline__ v16bf ldB16(const unsigned short* p) {
  u16x8 lo = *(const u16x8*)p;
  u16x8 hi = *(const u16x8*)(p + 16);
  u16x16 w = __builtin_shufflevector(lo, hi, 0,1,2,3,4,5,6,7,8,9,10,11,12,13,14,15);
  return __builtin_bit_cast(v16bf, w);
}

__device__ __forceinline__ v8f wmma_bf16(v16bf a, v16bf b, v8f c) {
  return __builtin_amdgcn_wmma_f32_16x16x32_bf16(false, a, false, b, (short)0, c,
                                                 false, false);
}

// ---- TDM: DMA a 2D [KCHUNK x 64] ushort tile into LDS, inserting a 16B pad
//      after every 128B row (-> 144B LDS row stride).  D# per ISA 8.3/8.4. ----
__device__ __forceinline__ void tdm_load_tile(unsigned lds_off, const void* gptr) {
  unsigned long long ga = (unsigned long long)gptr;
  u32x4 g0 = {0u, 0u, 0u, 0u};
  g0[0] = 1u;                                         // count=1 (valid load D#)
  g0[1] = lds_off;                                    // lds_addr (bytes)
  g0[2] = (unsigned)(ga & 0xFFFFFFFFull);             // global_addr[31:0]
  g0[3] = (unsigned)((ga >> 32) & 0x01FFFFFFull)      // global_addr[56:32]
          | 0x80000000u;                              // type=2 ("image")
  i32x8 g1 = {0, 0, 0, 0, 0, 0, 0, 0};
  g1[0] = (1 << 16)                                   // data_size = 2 bytes
        | (1 << 20)                                   // pad_enable
        | (4 << 22)                                   // pad_interval: 32 DWORDs (128B)
        | (3 << 25);                                  // pad_amount: 4 DWORDs (16B)
  g1[1] = (DDIM & 0xFFFF) << 16;                      // tensor_dim0 lo16 = 64
  g1[2] = (KCHUNK & 0xFFFF) << 16;                    // dim0 hi=0 | tensor_dim1 lo16
  g1[3] = (DDIM << 16);                               // dim1 hi=0 | tile_dim0 = 64
  g1[4] = KCHUNK;                                     // tile_dim1 = 128, tile_dim2 = 0
  g1[5] = DDIM;                                       // tensor_dim0_stride = 64 elems
  i32x4 gz  = {0, 0, 0, 0};                           // 2D: groups 2/3 unused
  i32x8 gz8 = {0, 0, 0, 0, 0, 0, 0, 0};               // extra group (clang-23 form)
  __builtin_amdgcn_tensor_load_to_lds(g0, g1, gz, gz, gz8, 0);
}

// ---- Pre-pass: codebook -> bf16 hi/lo planes + 0.5*||e||^2 (runs once) ----
__global__ __launch_bounds__(256) void vq_prep_codebook(
    const float* __restrict__ cb, unsigned short* __restrict__ wh,
    unsigned short* __restrict__ wl, float* __restrict__ wsq, int K) {
  const int code = blockIdx.x * 256 + threadIdx.x;
  if (code >= K) return;
  const float4* cp = (const float4*)(cb + (size_t)code * DDIM);
  unsigned short* ph = wh + (size_t)code * DDIM;
  unsigned short* pl = wl + (size_t)code * DDIM;
  float s = 0.f;
#pragma unroll
  for (int q = 0; q < 16; ++q) {
    float4 v = cp[q];
    float x[4] = {v.x, v.y, v.z, v.w};
    u16x4 hv, lv;
#pragma unroll
    for (int c = 0; c < 4; ++c) {
      unsigned short h = f2bf_rne(x[c]);
      hv[c] = h;
      lv[c] = f2bf_rne(x[c] - bf2f(h));
      s = fmaf(x[c], x[c], s);
    }
    *(u16x4*)(ph + q * 4) = hv;
    *(u16x4*)(pl + q * 4) = lv;
  }
  wsq[code] = 0.5f * s;
}

__global__ __launch_bounds__(256) void vq_argmin_wmma(
    const float* __restrict__ z, const float* __restrict__ cb,
    const unsigned short* __restrict__ wh, const unsigned short* __restrict__ wl,
    const float* __restrict__ wsq,
    float* __restrict__ out0, float* __restrict__ out1, int N, int K) {
  __shared__ unsigned short s_eh[2][KCHUNK * LDS_STRIDE];  // double-buffered hi
  __shared__ unsigned short s_el[2][KCHUNK * LDS_STRIDE];  // double-buffered lo
  __shared__ int            s_idx[ROWS_PER_BLOCK];

  const int tid  = threadIdx.x;
  const int lane = tid & 31;
  const int wv   = tid >> 5;
  const int row0 = blockIdx.x * ROWS_PER_BLOCK + wv * ROWS_PER_WAVE;
  const int m16  = lane & 15;
  const int off0 = (lane < 16) ? 0 : 8;
  if (row0 >= N) { /* grid sized exactly */ }

  const unsigned eh_off[2] = {(unsigned)(size_t)(void*)&s_eh[0][0],
                              (unsigned)(size_t)(void*)&s_eh[1][0]};
  const unsigned el_off[2] = {(unsigned)(size_t)(void*)&s_el[0][0],
                              (unsigned)(size_t)(void*)&s_el[1][0]};

  // ---- A fragments: bf16 hi/lo split of (-z) for this wave's 16 rows ----
  v16bf a_h[2], a_l[2];
  {
    const float* zr = z + (size_t)(row0 + m16) * DDIM;
#pragma unroll
    for (int s = 0; s < 2; ++s) {
      const float4* pA = (const float4*)(zr + 32 * s + off0);
      const float4* pB = (const float4*)(zr + 32 * s + off0 + 16);
      float4 v0 = pA[0], v1 = pA[1], v2 = pB[0], v3 = pB[1];
      float f[16] = {v0.x,v0.y,v0.z,v0.w, v1.x,v1.y,v1.z,v1.w,
                     v2.x,v2.y,v2.z,v2.w, v3.x,v3.y,v3.z,v3.w};
      u16x16 hb, lb;
#pragma unroll
      for (int i = 0; i < 16; ++i) {
        float y = -f[i];                       // fold the -1 of (-z.e) into A
        unsigned short h = f2bf_rne(y);
        hb[i] = h;
        lb[i] = f2bf_rne(y - bf2f(h));
      }
      a_h[s] = __builtin_bit_cast(v16bf, hb);
      a_l[s] = __builtin_bit_cast(v16bf, lb);
    }
  }

  float rmin[8];
  int   ridx[8];
#pragma unroll
  for (int j = 0; j < 8; ++j) { rmin[j] = 3.4028235e38f; ridx[j] = 0; }

  const int nChunks = K / KCHUNK;

  // Prologue: TDM-prefetch chunk 0 (wave 0 drives the DMA; EXEC-independent op)
  if (tid < 32) {
    tdm_load_tile(eh_off[0], wh);
    tdm_load_tile(el_off[0], wl);
  }

  for (int ch = 0; ch < nChunks; ++ch) {
    const int buf = ch & 1;
    __syncthreads();   // everyone done reading buffer buf^1 -> safe to refill it
    if (tid < 32) {
      if (ch + 1 < nChunks) {
        const size_t goff = (size_t)(ch + 1) * KCHUNK * DDIM;
        tdm_load_tile(eh_off[buf ^ 1], wh + goff);
        tdm_load_tile(el_off[buf ^ 1], wl + goff);
        __builtin_amdgcn_s_wait_tensorcnt(2);  // chunk ch landed; ch+1 in flight
      } else {
        __builtin_amdgcn_s_wait_tensorcnt(0);
      }
    }
    __syncthreads();   // publish chunk ch

    // Prefetch this chunk's 8 per-tile bias values into registers (one wait,
    // far from use) so the WMMA chains never stall on a scalar load.
    float sqs[8];
#pragma unroll
    for (int t = 0; t < 8; ++t) sqs[t] = wsq[ch * KCHUNK + t * 16 + m16];

    const unsigned short* behp = &s_eh[buf][0];
    const unsigned short* belp = &s_el[buf][0];
    const int koff = (lane < 16) ? 0 : 8;
#pragma unroll
    for (int t = 0; t < 8; ++t) {
      const int ccol = t * 16 + m16;
      const unsigned short* ph = behp + ccol * LDS_STRIDE + koff;
      const unsigned short* pl = belp + ccol * LDS_STRIDE + koff;
      v16bf bh0 = ldB16(ph);
      v16bf bh1 = ldB16(ph + 32);
      v16bf bl0 = ldB16(pl);
      v16bf bl1 = ldB16(pl + 32);
      v8f acc = {0.f, 0.f, 0.f, 0.f, 0.f, 0.f, 0.f, 0.f};  // inline-0 C operand
      acc = wmma_bf16(a_h[0], bh0, acc);       // acc = -z.e (hi/lo split)
      acc = wmma_bf16(a_l[0], bh0, acc);
      acc = wmma_bf16(a_h[0], bl0, acc);
      acc = wmma_bf16(a_h[1], bh1, acc);
      acc = wmma_bf16(a_l[1], bh1, acc);
      acc = wmma_bf16(a_h[1], bl1, acc);
      const int code = ch * KCHUNK + ccol;
#pragma unroll
      for (int j = 0; j < 8; ++j) {
        float sc = sqs[t] + acc[j];            // score = 0.5||e||^2 - z.e
        if (sc < rmin[j]) { rmin[j] = sc; ridx[j] = code; }  // first-min
      }
    }

    // Software-pipeline request: B-fragment DS reads of tile t+1 issue under
    // the WMMAs of tile t; score/argmin VALU trails. Keeps 2 B-buffers and
    // <=2 accumulators live instead of serializing on s_wait_dscnt 0.
    __builtin_amdgcn_sched_group_barrier(SGB_DSREAD, 8, 0);      // tile 0 loads
#pragma unroll
    for (int t = 0; t < 8; ++t) {
      __builtin_amdgcn_sched_group_barrier(SGB_MFMA, 6, 0);      // WMMAs of t
      if (t < 7)
        __builtin_amdgcn_sched_group_barrier(SGB_DSREAD, 8, 0);  // loads of t+1
      __builtin_amdgcn_sched_group_barrier(SGB_VALU, 32, 0);     // scores of t
    }
  }

  // ---- Per-row argmin across the 16 lanes of each half-wave ----
#pragma unroll
  for (int m = 1; m <= 8; m <<= 1) {
#pragma unroll
    for (int j = 0; j < 8; ++j) {
      float ov = __shfl_xor(rmin[j], m, 32);
      int   oi = __shfl_xor(ridx[j], m, 32);
      if (ov < rmin[j] || (ov == rmin[j] && oi < ridx[j])) { rmin[j] = ov; ridx[j] = oi; }
    }
  }
#pragma unroll
  for (int j = 0; j < 8; ++j) {
    if (lane == j)      s_idx[wv * 16 + j]     = ridx[j];
    if (lane == j + 16) s_idx[wv * 16 + 8 + j] = ridx[j];
  }
  // Same wave consumes its own slice; DS dependency wait is compiler-enforced.

  // ---- Gather fp32 codebook rows (L2-resident) and write both outputs ----
  const int r  = lane >> 1;
  const int hh = lane & 1;
  const int best = s_idx[wv * 16 + r];
  const float4* src = (const float4*)(cb + (size_t)best * DDIM + hh * 32);
  float4* d0 = (float4*)(out0 + (size_t)(row0 + r) * DDIM + hh * 32);
  float4* d1 = (float4*)(out1 + (size_t)(row0 + r) * DDIM + hh * 32);
#pragma unroll
  for (int q = 0; q < 8; ++q) {
    float4 v = src[q];
    d0[q] = v;
    d1[q] = v;
  }
}

extern "C" void kernel_launch(void* const* d_in, const int* in_sizes, int n_in,
                              void* d_out, int out_size, void* d_ws, size_t ws_size,
                              hipStream_t stream) {
  (void)n_in; (void)out_size; (void)ws_size;
  const float* z  = (const float*)d_in[0];
  const float* cb = (const float*)d_in[1];
  float* out = (float*)d_out;
  const int N = in_sizes[0] / DDIM;   // 131072
  const int K = in_sizes[1] / DDIM;   // 512

  // Workspace layout: [K*D ushort hi][K*D ushort lo][K float 0.5*||e||^2]
  unsigned short* wh  = (unsigned short*)d_ws;
  unsigned short* wl  = wh + (size_t)K * DDIM;
  float*          wsq = (float*)(wl + (size_t)K * DDIM);

  vq_prep_codebook<<<(K + 255) / 256, 256, 0, stream>>>(cb, wh, wl, wsq, K);

  float* out0 = out;
  float* out1 = out + (size_t)N * DDIM;
  vq_argmin_wmma<<<N / ROWS_PER_BLOCK, 256, 0, stream>>>(z, cb, wh, wl, wsq,
                                                         out0, out1, N, K);
}
